// patient_cls_88931592831280
// MI455X (gfx1250) — compile-verified, matrix operand order checked by reading, not run on hardware
//
#include <hip/hip_runtime.h>
#include <stdint.h>

#define N_NODES 201
#define M_PAD   208                            // 13 * 16, zero-padded rows in LDS
#define F_NODE  256
#define H_ENC   256
#define D_ENC   128
#define N_CLS   3
#define N_DEMO  6
#define EDGE_NUM 500
#define ROW_F32 (N_NODES * F_NODE + N_DEMO)   // 51462 floats per patient row
#define AFF_STRIDE 208

// ---- dynamic LDS layout (bytes), all 16-aligned ----
#define P_OFF      0u                          // bf16 p[208][256]            106,496 B
#define BIG_OFF    106496u                     // staging / aff f32 / h f32   205,856 B
#define RNORM_OFF  312352u                     // f32[256]
#define DEMO_OFF   313376u                     // f32[8]
#define POOLED_OFF 313408u                     // f32[256]
#define PPART_OFF  314432u                     // f32[256]
#define ENC_OFF    315456u                     // f32[128]
#define EDGE_OFF   315968u                     // u32[504]
#define HIST_OFF   317984u                     // u32[2048]
#define CTRL_OFF   326176u                     // i32[8]
#define SMEM_TOTAL 326208u                     // <= 327,680 (320 KB WGP LDS)

typedef __attribute__((ext_vector_type(16))) __bf16 v16bf;
typedef __attribute__((ext_vector_type(8)))  float  v8f;

union BFr { unsigned u[8]; v16bf v; };

__device__ __forceinline__ unsigned short f2bf(float f) {
  unsigned u = __float_as_uint(f);
  unsigned r = ((u >> 16) & 1u) + 0x7FFFu;     // round-to-nearest-even
  return (unsigned short)((u + r) >> 16);
}

__device__ __forceinline__ unsigned mono_key(float f) {
  unsigned u = __float_as_uint(f);
  return (u & 0x80000000u) ? ~u : (u | 0x80000000u);
}

// CDNA5 async global -> LDS copy (ASYNCcnt tracked)
__device__ __forceinline__ void async_ld_b64(unsigned lds_off, unsigned long long gaddr) {
  asm volatile("global_load_async_to_lds_b64 %0, %1, off"
               :: "v"(lds_off), "v"(gaddr) : "memory");
}
__device__ __forceinline__ void wait_async() {
  asm volatile("s_wait_asynccnt 0" ::: "memory");
}

// A/B fragment (16-bit, 16x32 A-layout): lane<16 -> K {0..7,16..23}, lane>=16 -> +8.
// Rows are pre-padded with zeros to M_PAD, so loads are unconditional (no cndmask).
__device__ __forceinline__ v16bf frag_rows(const unsigned short* base, int rowbase,
                                           int kbase, int lane) {
  BFr f;
  int half = lane >> 4;
  int row = rowbase + (lane & 15);
  const unsigned short* rp = base + row * F_NODE;
#pragma unroll
  for (int v = 0; v < 8; ++v) {
    int k0 = kbase + ((v < 4) ? (2 * v) : (8 + 2 * v)) + (half ? 8 : 0);
    f.u[v] = *(const unsigned*)(rp + k0);
  }
  return f.v;
}

// B fragment for W1 (K x N, row-major 256x256). Prefer pre-transposed bf16 W1t in ws.
__device__ __forceinline__ v16bf frag_w1(const float* __restrict__ W1,
                                         const unsigned short* __restrict__ W1t,
                                         int kbase, int nbase, int lane) {
  BFr f;
  int half = lane >> 4;
  int n = nbase + (lane & 15);
#pragma unroll
  for (int v = 0; v < 8; ++v) {
    int k0 = kbase + ((v < 4) ? (2 * v) : (8 + 2 * v)) + (half ? 8 : 0);
    if (W1t) {
      f.u[v] = *(const unsigned*)(W1t + n * F_NODE + k0);
    } else {
      unsigned lo = f2bf(W1[(size_t)k0 * H_ENC + n]);
      unsigned hi = f2bf(W1[(size_t)(k0 + 1) * H_ENC + n]);
      f.u[v] = lo | (hi << 16);
    }
  }
  return f.v;
}

__global__ void w1t_kernel(const float* __restrict__ W1, unsigned short* __restrict__ W1t) {
  int n = blockIdx.x, k = threadIdx.x;
  W1t[n * F_NODE + k] = f2bf(W1[(size_t)k * H_ENC + n]);
}

__global__ __launch_bounds__(256, 1)
void patient_kernel(const float* __restrict__ x,
                    const float* __restrict__ W1,
                    const unsigned short* __restrict__ W1t,
                    const float* __restrict__ Wout,
                    const float* __restrict__ b_out,
                    const float* __restrict__ W_fc,
                    const float* __restrict__ b_fc,
                    float* __restrict__ out) {
  extern __shared__ char smem[];
  unsigned short* pbf  = (unsigned short*)(smem + P_OFF);
  float*    big    = (float*)(smem + BIG_OFF);
  float*    rnorm  = (float*)(smem + RNORM_OFF);
  float*    demo   = (float*)(smem + DEMO_OFF);
  float*    pooled = (float*)(smem + POOLED_OFF);
  float*    ppart  = (float*)(smem + PPART_OFF);
  float*    enc    = (float*)(smem + ENC_OFF);
  unsigned* edge   = (unsigned*)(smem + EDGE_OFF);
  unsigned* hist   = (unsigned*)(smem + HIST_OFF);
  int*      ctrl   = (int*)(smem + CTRL_OFF);

  const int tid  = threadIdx.x;
  const int lane = tid & 31;
  const int wave = tid >> 5;
  const int blk  = blockIdx.x;
  const float* xrow = x + (size_t)blk * ROW_F32;

  // ---- Phase 0: async-stage the whole patient row (f32) into LDS 'big' ----
  {
    unsigned base_off = (unsigned)(uintptr_t)big;   // low 32 bits of flat shared ptr == LDS offset
    unsigned long long g = (unsigned long long)(uintptr_t)xrow;
    const int nch = (ROW_F32 * 4) / 8;              // 25731 x 8B chunks (row is 8B aligned)
    for (int c = tid; c < nch; c += 256)
      async_ld_b64(base_off + (unsigned)c * 8u, g + (unsigned long long)c * 8u);
    wait_async();
  }
  rnorm[tid] = 0.f;
  __syncthreads();

  // ---- Phase 1: f32 -> bf16 (+ zero pad rows), row sum-squares, demo extract ----
  for (int idx = tid; idx < N_NODES * F_NODE; idx += 256) {
    float v = big[idx];
    pbf[idx] = f2bf(v);
    atomicAdd(&rnorm[idx >> 8], v * v);
  }
  for (int idx = N_NODES * F_NODE + tid; idx < M_PAD * F_NODE; idx += 256)
    pbf[idx] = (unsigned short)0;                   // zero the 7 pad rows
  if (tid < N_DEMO) demo[tid] = big[N_NODES * F_NODE + tid];
  __syncthreads();
  if (tid < N_NODES) rnorm[tid] = rsqrtf(rnorm[tid]);
  for (int i = tid; i < 2048; i += 256) hist[i] = 0u;
  if (tid < 8) ctrl[tid] = 0;
  __syncthreads();

  // ---- Phase 2: aff = (p p^T) * rnorm_i * rnorm_j  via WMMA bf16 -> f32 LDS ----
  for (int t = wave; t < 13 * 13; t += 8) {
    int ti = t / 13, tj = t - ti * 13;
    int ibase = ti * 16, jbase = tj * 16;
    v8f acc = {};
#pragma unroll
    for (int kk = 0; kk < 8; ++kk) {
      v16bf a = frag_rows(pbf, ibase, kk * 32, lane);
      v16bf b = frag_rows(pbf, jbase, kk * 32, lane);   // B tile = rows of p (p^T)
      acc = __builtin_amdgcn_wmma_f32_16x16x32_bf16(false, a, false, b,
                                                    (short)0, acc, false, false);
    }
    int n = lane & 15, mo = (lane >> 4) * 8;
#pragma unroll
    for (int r = 0; r < 8; ++r) {
      int i = ibase + mo + r, j = jbase + n;
      if (i < N_NODES && j < N_NODES)
        big[i * AFF_STRIDE + j] = acc[r] * rnorm[i] * rnorm[j];
    }
  }
  __syncthreads();

  // ---- Phase 3: top-500 via 2-level 11-bit histogram on monotonic keys ----
  for (int idx = tid; idx < N_NODES * N_NODES; idx += 256) {
    int i = idx / N_NODES, j = idx - i * N_NODES;
    unsigned key = mono_key(big[i * AFF_STRIDE + j]);
    atomicAdd(&hist[key >> 21], 1u);
  }
  __syncthreads();
  if (tid == 0) {
    unsigned c = 0; int b1 = 0; unsigned above = 0;
    for (int b = 2047; b >= 0; --b) {
      unsigned nc = c + hist[b];
      if (nc >= EDGE_NUM) { b1 = b; above = c; break; }
      c = nc;
    }
    ctrl[0] = b1; ctrl[1] = (int)above;
  }
  __syncthreads();
  for (int i = tid; i < 2048; i += 256) hist[i] = 0u;
  __syncthreads();
  {
    int b1 = ctrl[0];
    for (int idx = tid; idx < N_NODES * N_NODES; idx += 256) {
      int i = idx / N_NODES, j = idx - i * N_NODES;
      unsigned key = mono_key(big[i * AFF_STRIDE + j]);
      if ((int)(key >> 21) == b1) atomicAdd(&hist[(key >> 10) & 2047u], 1u);
    }
  }
  __syncthreads();
  if (tid == 0) {
    unsigned c = (unsigned)ctrl[1]; int b2 = 0; unsigned above2 = c;
    for (int b = 2047; b >= 0; --b) {
      unsigned nc = c + hist[b];
      if (nc >= EDGE_NUM) { b2 = b; above2 = c; break; }
      c = nc;
    }
    ctrl[2] = (ctrl[0] << 11) | b2;          // 22-bit threshold key prefix
    ctrl[3] = (int)above2;                    // count strictly above
    ctrl[4] = EDGE_NUM - (int)above2;         // ties to take
    ctrl[5] = 0; ctrl[6] = 0;
  }
  __syncthreads();
  {
    int T = ctrl[2], cA = ctrl[3], need = ctrl[4];
    for (int idx = tid; idx < N_NODES * N_NODES; idx += 256) {
      int i = idx / N_NODES, j = idx - i * N_NODES;
      unsigned key = mono_key(big[i * AFF_STRIDE + j]);
      int k22 = (int)(key >> 10);
      if (k22 > T) {
        int s = atomicAdd(&ctrl[5], 1);
        edge[s] = ((unsigned)i << 16) | (unsigned)j;
      } else if (k22 == T) {
        int s = atomicAdd(&ctrl[6], 1);
        if (s < need) edge[cA + s] = ((unsigned)i << 16) | (unsigned)j;
      }
    }
  }
  __syncthreads();

  // ---- Phase 4: h = p @ W1 (WMMA), f32 h overwrites aff region ----
  for (int t = wave; t < 13 * 16; t += 8) {
    int ti = t >> 4, tj = t & 15;
    int ibase = ti * 16, nbase = tj * 16;
    v8f acc = {};
#pragma unroll
    for (int kk = 0; kk < 8; ++kk) {
      v16bf a = frag_rows(pbf, ibase, kk * 32, lane);
      v16bf b = frag_w1(W1, W1t, kk * 32, nbase, lane);
      acc = __builtin_amdgcn_wmma_f32_16x16x32_bf16(false, a, false, b,
                                                    (short)0, acc, false, false);
    }
    int n = lane & 15, mo = (lane >> 4) * 8;
#pragma unroll
    for (int r = 0; r < 8; ++r) {
      int i = ibase + mo + r;
      if (i < N_NODES) big[i * F_NODE + nbase + n] = acc[r];
    }
  }
  __syncthreads();

  // ---- Phase 5: scatter-add agg (2 halves of 128 cols; agg reuses dead p region) ----
  float* agg = (float*)(smem + P_OFF);   // 201*128 f32 fits in old bf16 p region
  for (int hc = 0; hc < 2; ++hc) {
    for (int t = tid; t < N_NODES * 128; t += 256) agg[t] = 0.f;
    __syncthreads();
    for (int t = tid; t < EDGE_NUM * 128; t += 256) {
      int e = t >> 7, j = t & 127;
      unsigned rc = edge[e];
      int r = (int)(rc >> 16), c = (int)(rc & 0xFFFFu);
      atomicAdd(&agg[r * 128 + j], big[c * F_NODE + hc * 128 + j]);
    }
    __syncthreads();
    {
      int j = tid & 127, g = tid >> 7;
      float s = 0.f;
      for (int r = g; r < N_NODES; r += 2)
        s += fmaxf(big[r * F_NODE + hc * 128 + j] + agg[r * 128 + j], 0.f);
      ppart[g * 128 + j] = s;
    }
    __syncthreads();
    if (tid < 128)
      pooled[hc * 128 + tid] = (ppart[tid] + ppart[128 + tid]) * (1.0f / (float)N_NODES);
    __syncthreads();
  }

  // ---- Phase 6: enc = pooled @ Wout + b_out ; out = [enc,demo] @ W_fc + b_fc ----
  if (tid < D_ENC) {
    float s = b_out[tid];
    for (int k = 0; k < H_ENC; ++k) s += pooled[k] * Wout[k * D_ENC + tid];
    enc[tid] = s;
  }
  __syncthreads();
  if (tid < N_CLS) {
    float s = b_fc[tid];
    for (int k = 0; k < D_ENC; ++k) s += enc[k] * W_fc[k * N_CLS + tid];
    for (int d = 0; d < N_DEMO; ++d) s += demo[d] * W_fc[(D_ENC + d) * N_CLS + tid];
    out[(size_t)blk * N_CLS + tid] = s;
  }
}

extern "C" void kernel_launch(void* const* d_in, const int* in_sizes, int n_in,
                              void* d_out, int out_size, void* d_ws, size_t ws_size,
                              hipStream_t stream) {
  const float* x     = (const float*)d_in[0];
  const float* W1    = (const float*)d_in[1];
  const float* Wout  = (const float*)d_in[2];
  const float* b_out = (const float*)d_in[3];
  const float* W_fc  = (const float*)d_in[4];
  const float* b_fc  = (const float*)d_in[5];
  int B = in_sizes[0] / ROW_F32;

  unsigned short* W1t = nullptr;
  if (d_ws && ws_size >= (size_t)(F_NODE * H_ENC * 2)) {
    W1t = (unsigned short*)d_ws;
    hipLaunchKernelGGL(w1t_kernel, dim3(H_ENC), dim3(F_NODE), 0, stream, W1, W1t);
  }

  (void)hipFuncSetAttribute((const void*)patient_kernel,
                            hipFuncAttributeMaxDynamicSharedMemorySize,
                            (int)SMEM_TOTAL);
  hipLaunchKernelGGL(patient_kernel, dim3(B), dim3(256), SMEM_TOTAL, stream,
                     x, W1, W1t, Wout, b_out, W_fc, b_fc, (float*)d_out);
}
